// Block_37383395345129
// MI455X (gfx1250) — matrix operand order repaired and checked by reading, not compiled, not run on hardware
//
#include <hip/hip_runtime.h>
#include <hip/hip_bf16.h>

typedef _Float16 f16;
typedef __attribute__((ext_vector_type(16))) _Float16 v16h;
typedef __attribute__((ext_vector_type(8)))  float    v8f;

#define NB   8
#define NN   1024
#define NC   48
#define NH   8
#define HD   6
#define NTOK (NB * NN)      // 8192
#define NHID 24
#define NK   8              // GRID_SIZE + SPLINE_ORDER

// ---------------------------------------------------------------------------
// LayerNorm: one wave32 per token (C=48: lane handles elem L and L+32)
// ---------------------------------------------------------------------------
__global__ __launch_bounds__(256) void ln_kernel(const float* __restrict__ in,
                                                 const float* __restrict__ g,
                                                 const float* __restrict__ bta,
                                                 float* __restrict__ out, int ntok)
{
    int wid  = (blockIdx.x * blockDim.x + threadIdx.x) >> 5;
    int lane = threadIdx.x & 31;
    if (wid >= ntok) return;
    const float* xr = in + (size_t)wid * NC;
    float e0 = xr[lane];
    float e1 = (lane < 16) ? xr[32 + lane] : 0.f;
    float s = e0 + e1;
#pragma unroll
    for (int o = 16; o > 0; o >>= 1) s += __shfl_xor(s, o, 32);
    float mu = s * (1.f / 48.f);
    float d0 = e0 - mu;
    float d1 = (lane < 16) ? (e1 - mu) : 0.f;
    float v = d0 * d0 + d1 * d1;
#pragma unroll
    for (int o = 16; o > 0; o >>= 1) v += __shfl_xor(v, o, 32);
    float rinv = rsqrtf(v * (1.f / 48.f) + 1e-5f);
    float* orow = out + (size_t)wid * NC;
    orow[lane] = d0 * rinv * g[lane] + bta[lane];
    if (lane < 16) orow[32 + lane] = d1 * rinv * g[32 + lane] + bta[32 + lane];
}

// ---------------------------------------------------------------------------
// QKV projection: one thread per (token, j) ; scatter to (B,H,N,hd) buffers
// ---------------------------------------------------------------------------
__global__ __launch_bounds__(256) void qkv_kernel(const float* __restrict__ n1,
                                                  const float* __restrict__ w,
                                                  float* __restrict__ qb,
                                                  float* __restrict__ kb,
                                                  float* __restrict__ vb)
{
    int idx = blockIdx.x * blockDim.x + threadIdx.x;
    if (idx >= NTOK * 3 * NC) return;
    int tok = idx / (3 * NC), j = idx % (3 * NC);
    const float* xr = n1 + (size_t)tok * NC;
    const float* wr = w + (size_t)j * NC;
    float acc = 0.f;
#pragma unroll 8
    for (int c = 0; c < NC; ++c) acc += xr[c] * wr[c];
    int which = j / NC, rem = j % NC;
    int h = rem / HD, d = rem % HD;
    int b = tok >> 10, n = tok & 1023;
    float* dst = (which == 0) ? qb : (which == 1) ? kb : vb;
    dst[(((size_t)b * NH + h) * NN + n) * HD + d] = acc;
}

// ---------------------------------------------------------------------------
// Fused attention: 4 waves (128 threads) per (b, h, 16-query window).
//   Waves split the 64 key tiles of S = Q K^T (v_wmma_f32_16x16x32_f16,
//   hd=6 zero-padded), share the 16x1024 f32 score tile in LDS, split the
//   16 softmax rows (4/wave -> 4x outstanding HBM stores of attn_global),
//   and split P@V into 4 partial WMMA accumulations reduced through LDS.
//   Local (16-wide window) attention == diagonal 16x16 tile of S.
// ---------------------------------------------------------------------------
__global__ __launch_bounds__(128) void attn_kernel(const float* __restrict__ qb,
                                                   const float* __restrict__ kb,
                                                   const float* __restrict__ vb,
                                                   float* __restrict__ attn_out,
                                                   float* __restrict__ x_att)
{
    extern __shared__ float smem[];          // [16][1024] scores -> probs
    float* sloc  = smem + 16 * NN;           // [16][16] raw diagonal scores
    float* xloc  = sloc + 256;               // [16][8]  local outputs
    float* opart = xloc + 128;               // [4][32][8] P@V partials

    const int tid  = threadIdx.x;
    const int wv   = tid >> 5;               // wave 0..3
    const int lane = tid & 31;
    const int blk  = blockIdx.x;             // 0..4095
    const int qwin = blk & 63;
    const int bh   = blk >> 6;               // b*8 + h
    const int b    = bh >> 3, h = bh & 7;

    const float scale = rsqrtf((float)HD);
    const float* qbase = qb + (size_t)bh * NN * HD;
    const float* kbase = kb + (size_t)bh * NN * HD;
    const float* vbase = vb + (size_t)bh * NN * HD;

    const int rb  = (lane >> 4) * 8;         // C/D row base (lanes 16-31 -> M 8..15)
    const int nlo = lane & 15;

    // ---- A fragment: Q tile (16x32 f16, only K=0..5 nonzero, lanes 0-15) ----
    v16h afrag = {};
    if (lane < 16) {
        const float* qp = qbase + (size_t)(qwin * 16 + lane) * HD;
#pragma unroll
        for (int d = 0; d < HD; ++d) afrag[d] = (f16)qp[d];
    }

    // ---- S = Q K^T : each wave owns 16 of the 64 key tiles ----
    for (int i = 0; i < 16; ++i) {
        const int kt = wv * 16 + i;
        v16h bfrag = {};
        if (lane < 16) {
            const float* kp = kbase + (size_t)(kt * 16 + lane) * HD;
#pragma unroll
            for (int d = 0; d < HD; ++d) bfrag[d] = (f16)kp[d];
        }
        v8f c = {};
        c = __builtin_amdgcn_wmma_f32_16x16x32_f16(false, afrag, false, bfrag,
                                                   (short)0, c, false, false);
#pragma unroll
        for (int r = 0; r < 8; ++r)
            smem[(rb + r) * NN + kt * 16 + nlo] = c[r] * scale;
    }
    __syncthreads();

    // ---- stash raw diagonal tile for local attention ----
    for (int i = tid; i < 256; i += 128)
        sloc[i] = smem[(i >> 4) * NN + qwin * 16 + (i & 15)];
    __syncthreads();

    // ---- global softmax: wave wv owns rows 4*wv .. 4*wv+3 ----
    float* arow_base = attn_out + ((size_t)bh * NN + (size_t)qwin * 16) * NN;
    for (int i = 0; i < 4; ++i) {
        const int m = wv * 4 + i;
        float* srow = smem + m * NN;
        float mx = -3.4e38f;
        for (int j = lane; j < NN; j += 32) mx = fmaxf(mx, srow[j]);
#pragma unroll
        for (int o = 16; o > 0; o >>= 1) mx = fmaxf(mx, __shfl_xor(mx, o, 32));
        float sum = 0.f;
        for (int j = lane; j < NN; j += 32) {
            float e = __expf(srow[j] - mx);
            srow[j] = e;
            sum += e;
        }
#pragma unroll
        for (int o = 16; o > 0; o >>= 1) sum += __shfl_xor(sum, o, 32);
        float inv = 1.0f / sum;
        float* grow = arow_base + (size_t)m * NN;
        for (int j = lane; j < NN; j += 32) {
            float p = srow[j] * inv;
            srow[j] = p;
            grow[j] = p;              // attn_global output (coalesced)
        }
    }
    __syncthreads();

    // ---- local window attention (diagonal 16x16), VALU, threads 0-15 ----
    if (tid < 16) {
        const int m = tid;
        float mx = -3.4e38f;
#pragma unroll
        for (int j = 0; j < 16; ++j) mx = fmaxf(mx, sloc[m * 16 + j]);
        float p[16], sum = 0.f;
#pragma unroll
        for (int j = 0; j < 16; ++j) { p[j] = __expf(sloc[m * 16 + j] - mx); sum += p[j]; }
        float inv = 1.f / sum;
        const float* vwin = vbase + (size_t)(qwin * 16) * HD;
#pragma unroll
        for (int d = 0; d < HD; ++d) {
            float acc = 0.f;
#pragma unroll
            for (int j = 0; j < 16; ++j) acc += p[j] * vwin[j * HD + d];
            xloc[m * 8 + d] = acc * inv;
        }
    }

    // ---- O = P @ V : wave wv owns keys [wv*256, wv*256+256) => 8 WMMAs ----
    v8f oacc = {};
    for (int i = 0; i < 8; ++i) {
        const int k0 = (wv * 8 + i) * 32;
        v16h pa;
        {
            const int m    = lane & 15;
            const int koff = k0 + ((lane >> 4) << 3);
            const float* prow = smem + m * NN + koff;
#pragma unroll
            for (int j = 0; j < 8; ++j) pa[j] = (f16)prow[j];
#pragma unroll
            for (int j = 0; j < 8; ++j) pa[8 + j] = (f16)prow[16 + j];
        }
        v16h vf = {};
        {
            const int n  = lane & 15;
            const int kk = k0 + ((lane >> 4) << 3);
            if (n < HD) {
                const float* vp = vbase + (size_t)kk * HD + n;
#pragma unroll
                for (int j = 0; j < 8; ++j) vf[j] = (f16)vp[j * HD];
#pragma unroll
                for (int j = 0; j < 8; ++j) vf[8 + j] = (f16)vp[(16 + j) * HD];
            }
        }
        oacc = __builtin_amdgcn_wmma_f32_16x16x32_f16(false, pa, false, vf,
                                                      (short)0, oacc, false, false);
    }
#pragma unroll
    for (int r = 0; r < 8; ++r) opart[(wv * 32 + lane) * 8 + r] = oacc[r];
    __syncthreads();

    // ---- wave 0 reduces 4 partials, adds local, writes x_att (B,N,H*hd) ----
    if (wv == 0 && nlo < HD) {
#pragma unroll
        for (int r = 0; r < 8; ++r) {
            float acc = xloc[(rb + r) * 8 + nlo];
#pragma unroll
            for (int w2 = 0; w2 < 4; ++w2) acc += opart[(w2 * 32 + lane) * 8 + r];
            const int tok = qwin * 16 + rb + r;
            x_att[((size_t)b * NN + tok) * NC + h * HD + nlo] = acc;
        }
    }
}

// ---------------------------------------------------------------------------
// proj + residual: x1 = x + x_att @ w_proj^T + b_proj
// ---------------------------------------------------------------------------
__global__ __launch_bounds__(256) void proj_kernel(const float* __restrict__ x,
                                                   const float* __restrict__ xatt,
                                                   const float* __restrict__ w,
                                                   const float* __restrict__ bias,
                                                   float* __restrict__ x1)
{
    int idx = blockIdx.x * blockDim.x + threadIdx.x;
    if (idx >= NTOK * NC) return;
    int tok = idx / NC, o = idx % NC;
    const float* ar = xatt + (size_t)tok * NC;
    const float* wr = w + (size_t)o * NC;
    float acc = bias[o];
#pragma unroll 8
    for (int c = 0; c < NC; ++c) acc += ar[c] * wr[c];
    x1[idx] = x[idx] + acc;
}

// ---------------------------------------------------------------------------
// KAN B-spline basis (grid g[j] = -2.2 + 0.4 j, j=0..11; cubic; 8 outputs)
// ---------------------------------------------------------------------------
__device__ inline void kan_basis(float x, float* __restrict__ out)
{
    float bb[11];
#pragma unroll
    for (int j = 0; j < 11; ++j) {
        float gl = -2.2f + 0.4f * j;
        bb[j] = (x >= gl && x < gl + 0.4f) ? 1.f : 0.f;
    }
#pragma unroll
    for (int k = 1; k <= 3; ++k) {
        float invk = 1.0f / (0.4f * k);
#pragma unroll
        for (int j = 0; j < 11 - k; ++j) {
            float gj   = -2.2f + 0.4f * j;
            float gjk1 = gj + 0.4f * (k + 1);
            bb[j] = (x - gj) * invk * bb[j] + (gjk1 - x) * invk * bb[j + 1];
        }
    }
#pragma unroll
    for (int j = 0; j < NK; ++j) out[j] = bb[j];
}

// ---------------------------------------------------------------------------
// KAN MLP: one 64-thread block per token; basis tables staged in LDS
// ---------------------------------------------------------------------------
__global__ __launch_bounds__(64) void kan_kernel(const float* __restrict__ n2,
                                                 const float* __restrict__ x1,
                                                 const float* __restrict__ bw1,
                                                 const float* __restrict__ sw1,
                                                 const float* __restrict__ ss1,
                                                 const float* __restrict__ bw2,
                                                 const float* __restrict__ sw2,
                                                 const float* __restrict__ ss2,
                                                 float* __restrict__ x2)
{
    __shared__ float bas1[NC][NK];
    __shared__ float act1[NC];
    __shared__ float hid[NHID];
    __shared__ float bas2[NHID][NK];
    __shared__ float act2[NHID];

    int tok = blockIdx.x;
    int t   = threadIdx.x;
    const float* xr = n2 + (size_t)tok * NC;

    if (t < NC) {
        float xv = xr[t];
        kan_basis(xv, &bas1[t][0]);
        act1[t] = xv / (1.f + __expf(-xv));   // silu
    }
    __syncthreads();

    if (t < NHID) {
        float acc = 0.f;
        for (int c = 0; c < NC; ++c) {
            acc += act1[c] * bw1[t * NC + c];
            const float* swr = sw1 + (size_t)(t * NC + c) * NK;
            float sp = 0.f;
#pragma unroll
            for (int k = 0; k < NK; ++k) sp += bas1[c][k] * swr[k];
            acc += sp * ss1[t * NC + c];
        }
        hid[t] = acc;
        kan_basis(acc, &bas2[t][0]);
        act2[t] = acc / (1.f + __expf(-acc));
    }
    __syncthreads();

    if (t < NC) {
        float acc = 0.f;
        for (int i = 0; i < NHID; ++i) {
            acc += act2[i] * bw2[t * NHID + i];
            const float* swr = sw2 + (size_t)(t * NHID + i) * NK;
            float sp = 0.f;
#pragma unroll
            for (int k = 0; k < NK; ++k) sp += bas2[i][k] * swr[k];
            acc += sp * ss2[t * NHID + i];
        }
        x2[(size_t)tok * NC + t] = x1[(size_t)tok * NC + t] + acc;
    }
}

// ---------------------------------------------------------------------------
// conv path: conv1 (48->128, k=3 same) then conv3 (128->48), on original x
// ---------------------------------------------------------------------------
__global__ __launch_bounds__(256) void conv1_kernel(const float* __restrict__ x,
                                                    const float* __restrict__ w,
                                                    const float* __restrict__ bias,
                                                    float* __restrict__ c1)
{
    int idx = blockIdx.x * blockDim.x + threadIdx.x;   // (b,o,t) = 8*128*1024
    if (idx >= NB * 128 * NN) return;
    int t = idx & 1023;
    int o = (idx >> 10) & 127;
    int b = idx >> 17;
    float acc = bias[o];
#pragma unroll
    for (int kk = 0; kk < 3; ++kk) {
        int tt = t + kk - 1;
        if (tt < 0 || tt >= NN) continue;
        const float* xr = x + ((size_t)b * NN + tt) * NC;
        const float* wr = w + (size_t)(o * NC) * 3 + kk;
#pragma unroll 8
        for (int i = 0; i < NC; ++i) acc += xr[i] * wr[i * 3];
    }
    c1[idx] = acc;
}

__global__ __launch_bounds__(256) void conv3_kernel(const float* __restrict__ c1,
                                                    const float* __restrict__ w,
                                                    const float* __restrict__ bias,
                                                    float* __restrict__ h2)
{
    int idx = blockIdx.x * blockDim.x + threadIdx.x;   // (b,o,t) = 8*48*1024
    if (idx >= NB * NC * NN) return;
    int t = idx % NN;
    int o = (idx / NN) % NC;
    int b = idx / (NC * NN);
    float acc = bias[o];
#pragma unroll
    for (int kk = 0; kk < 3; ++kk) {
        int tt = t + kk - 1;
        if (tt < 0 || tt >= NN) continue;
        const float* cr = c1 + (size_t)b * 128 * NN + tt;
        const float* wr = w + (size_t)(o * 128) * 3 + kk;
#pragma unroll 8
        for (int i = 0; i < 128; ++i) acc += cr[i * NN] * wr[i * 3];
    }
    h2[((size_t)b * NN + t) * NC + o] = acc;           // token-major for LN reuse
}

__global__ __launch_bounds__(256) void final_kernel(const float* __restrict__ lnh,
                                                    const float* __restrict__ x2,
                                                    const float* __restrict__ sc,
                                                    float* __restrict__ out)
{
    int idx = blockIdx.x * blockDim.x + threadIdx.x;
    if (idx >= NTOK * NC) return;
    out[idx] = sc[0] * lnh[idx] + x2[idx];
}

// ---------------------------------------------------------------------------
extern "C" void kernel_launch(void* const* d_in, const int* in_sizes, int n_in,
                              void* d_out, int out_size, void* d_ws, size_t ws_size,
                              hipStream_t stream)
{
    const float* x       = (const float*)d_in[0];
    const float* g1      = (const float*)d_in[1];
    const float* b1      = (const float*)d_in[2];
    const float* w_qkv   = (const float*)d_in[3];
    const float* w_proj  = (const float*)d_in[4];
    const float* b_proj  = (const float*)d_in[5];
    const float* g2      = (const float*)d_in[6];
    const float* b2      = (const float*)d_in[7];
    const float* bw1     = (const float*)d_in[8];
    const float* sw1     = (const float*)d_in[9];
    const float* ss1     = (const float*)d_in[10];
    const float* bw2     = (const float*)d_in[11];
    const float* sw2     = (const float*)d_in[12];
    const float* ss2     = (const float*)d_in[13];
    const float* conv1_w = (const float*)d_in[14];
    const float* conv1_b = (const float*)d_in[15];
    const float* conv3_w = (const float*)d_in[16];
    const float* conv3_b = (const float*)d_in[17];
    const float* sc      = (const float*)d_in[18];

    float* out      = (float*)d_out;              // (8,1024,48)
    float* attn_out = (float*)d_out + NTOK * NC;  // (8,8,1024,1024)

    const size_t SZ = (size_t)NTOK * NC;          // 393216
    float* ws   = (float*)d_ws;
    size_t off  = 0;
    float* n1   = ws + off; off += SZ;
    float* qb   = ws + off; off += SZ;
    float* kb   = ws + off; off += SZ;
    float* vb   = ws + off; off += SZ;
    float* xatt = ws + off; off += SZ;
    float* x1   = ws + off; off += SZ;
    float* n2b  = ws + off; off += SZ;
    float* x2   = ws + off; off += SZ;
    float* c1   = ws + off; off += (size_t)NB * 128 * NN;
    float* h2   = ws + off; off += SZ;
    float* lnh  = ws + off; off += SZ;

    // 1. n1 = LN(x)
    ln_kernel<<<NTOK / 8, 256, 0, stream>>>(x, g1, b1, n1, NTOK);
    // 2. q,k,v
    qkv_kernel<<<(NTOK * 3 * NC) / 256, 256, 0, stream>>>(n1, w_qkv, qb, kb, vb);
    // 3. fused global+local attention (WMMA), writes attn_global + x_att
    const int attn_lds = (16 * NN + 256 + 128 + 4 * 32 * 8) * sizeof(float); // 71168 B
    attn_kernel<<<NB * NH * (NN / 16), 128, attn_lds, stream>>>(qb, kb, vb, attn_out, xatt);
    // 4. x1 = x + proj(x_att)
    proj_kernel<<<(NTOK * NC) / 256, 256, 0, stream>>>(x, xatt, w_proj, b_proj, x1);
    // 5. n2 = LN(x1)
    ln_kernel<<<NTOK / 8, 256, 0, stream>>>(x1, g2, b2, n2b, NTOK);
    // 6. x2 = x1 + KAN(n2)
    kan_kernel<<<NTOK, 64, 0, stream>>>(n2b, x1, bw1, sw1, ss1, bw2, sw2, ss2, x2);
    // 7-8. conv path on original x
    conv1_kernel<<<(NB * 128 * NN) / 256, 256, 0, stream>>>(x, conv1_w, conv1_b, c1);
    conv3_kernel<<<(NB * NC * NN) / 256, 256, 0, stream>>>(c1, conv3_w, conv3_b, h2);
    // 9. LN(h) with g1,b1 (Block reuses norm1)
    ln_kernel<<<NTOK / 8, 256, 0, stream>>>(h2, g1, b1, lnh, NTOK);
    // 10. out = sc*lnh + x2
    final_kernel<<<(NTOK * NC) / 256, 256, 0, stream>>>(lnh, x2, sc, out);
}
